// ScaledDotProductionAttention_33698313404452
// MI455X (gfx1250) — matrix operand order repaired and checked by reading, not compiled, not run on hardware
//
#include <hip/hip_runtime.h>
#include <stdint.h>

#define B_ 16
#define S_ 2048
#define D_ 128
#define SCALE 0.08838834764831845f  /* 1/sqrt(128) */

typedef __attribute__((ext_vector_type(16))) _Float16 v16h;
typedef __attribute__((ext_vector_type(8)))  _Float16 v8h;
typedef __attribute__((ext_vector_type(8)))  float    v8f;
typedef __attribute__((ext_vector_type(4)))  unsigned int v4u;
typedef __attribute__((ext_vector_type(8)))  int          v8i;
typedef __attribute__((ext_vector_type(4)))  int          v4i;

union HFrag { v16h v; v8h h[2]; };

// TDM-staged 32x128h tile: 32 rows x 256B, padded by TDM to 272B stride (16B
// pad per 256B) -> 16B-aligned ds_load_b128 fragments, banks 4*ln.
#define ROW_STRIDE 272
#define BUF_BYTES  8704  /* 32 * 272 */

// ---- Tensor Data Mover: 1-D contiguous 8KB load (1024 x 8B) with LDS padding.
// D# per CDNA5 ISA ch.8: group0 {count=1, lds_addr, global_addr, type=2},
// group1 {data_size=3(8B), pad_enable, pad_interval=5(256B), pad_amount=3(16B),
//         tensor_dim0=1024, tensor_dim1=1, tile_dim0=1024, tile_dim1=1,
//         tensor_dim0_stride=1024}, groups 2/3 zero (<=2D form).
// clang-23 toolchain: 6-arg builtin (g0 v4u, g1 v8i, g2 v4i, g3 v4i, v8i, cpol).
__device__ __forceinline__ void tdm_load_tile(const _Float16* src, const void* lds_dst) {
    uint64_t ga = (uint64_t)(uintptr_t)src;
    v4u g0 = { 1u,                                   // count=1, user mode
               (unsigned int)(uintptr_t)lds_dst,     // LDS byte address
               (unsigned int)ga,                     // global_addr[31:0]
               (unsigned int)(ga >> 32) | (2u << 30) /* type=2 ("image") */ };
    v8i g1 = { (int)((3u << 16) | (1u << 20) | (5u << 22) | (3u << 25)),
               (int)(1024u << 16),   // tensor_dim0[15:0] in bits 63:48
               (int)(1u << 16),      // tensor_dim0[31:16]=0 | tensor_dim1 lo = 1
               (int)(1024u << 16),   // tensor_dim1 hi = 0 | tile_dim0 = 1024
               1,                    // tile_dim1 = 1, tile_dim2 = 0
               1024, 0, 0 };         // tensor_dim0_stride = 1024, dim1_stride = 0
    v4i gz4 = { 0, 0, 0, 0 };
    v8i gz8 = { 0, 0, 0, 0, 0, 0, 0, 0 };
    __builtin_amdgcn_tensor_load_to_lds(g0, g1, gz4, gz4, gz8, 0);
}

// ---------------- Pass 0a: convert Q, K to f16 ----------------
__global__ void cvt_qk_kernel(const float* __restrict__ q, const float* __restrict__ k,
                              _Float16* __restrict__ qh, _Float16* __restrict__ kh) {
    size_t i = (size_t)blockIdx.x * blockDim.x + threadIdx.x;
    qh[i] = (_Float16)q[i];
    kh[i] = (_Float16)k[i];
}

// ---------------- Pass 0b: V -> f16, transposed to [b][d][k] ----------------
__global__ void cvt_vt_kernel(const float* __restrict__ v, _Float16* __restrict__ vt) {
    size_t i = (size_t)blockIdx.x * blockDim.x + threadIdx.x;  // over B*D*S
    int k = (int)(i % S_);
    int d = (int)((i / S_) % D_);
    int b = (int)(i / ((size_t)S_ * D_));
    vt[i] = (_Float16)v[((size_t)b * S_ + k) * D_ + d];
}

// ---------------- Pass 1: Zinv[b][k] = 1 / sum_q exp(masked_score) ----------------
// Softmax is over the QUERY axis (axis=1 quirk): column sums of the S x S scores.
// Q tiles (shared by all 8 waves of the block) are staged by TDM, double-buffered.
__global__ void __launch_bounds__(256)
attn_stats_kernel(const _Float16* __restrict__ qh, const _Float16* __restrict__ kh,
                  const uint8_t* __restrict__ mask, float* __restrict__ zinv) {
    __shared__ __align__(16) unsigned char ldsQ[2][BUF_BYTES];

    const int wave = threadIdx.x >> 5;
    const int lane = threadIdx.x & 31;
    const int ln = lane & 15;
    const int h  = lane >> 4;
    const int b  = blockIdx.y;
    const int kbase = (blockIdx.x * 8 + wave) * 16;  // this wave's 16 key columns

    // Preload B fragments: 16 K-rows x 128 features, 4 chunks of K=32
    const _Float16* krow = kh + ((size_t)b * S_ + kbase + ln) * D_;
    v16h kb[4];
#pragma unroll
    for (int c = 0; c < 4; ++c)
        kb[c] = *(const v16h*)(krow + c * 32 + 16 * h);

    const uint8_t* mcol = mask + (size_t)b * S_ * S_ + kbase + ln;
    float colsum = 0.f;

    if (wave == 0)  // DMA of the first 32 Q rows
        tdm_load_tile(qh + (size_t)b * S_ * D_, ldsQ[0]);

    for (int qt2 = 0; qt2 < S_ / 32; ++qt2) {
        const unsigned char* qbuf = ldsQ[qt2 & 1];

        __builtin_amdgcn_s_wait_tensorcnt(0);  // wave0: current tile landed
        __syncthreads();                       // publish to all waves

        if (wave == 0 && qt2 + 1 < S_ / 32)    // prefetch next 32 Q rows
            tdm_load_tile(qh + ((size_t)b * S_ + (size_t)(qt2 + 1) * 32) * D_,
                          ldsQ[(qt2 + 1) & 1]);

#pragma unroll
        for (int st = 0; st < 2; ++st) {
            const int qbase = qt2 * 32 + st * 16;
            const unsigned char* qrow = qbuf + (size_t)(st * 16 + ln) * ROW_STRIDE;
            v8f acc = {};
#pragma unroll
            for (int c = 0; c < 4; ++c) {
                HFrag a;  // A frag: two 16B pieces of the staged Q row
                a.h[0] = *(const v8h*)(qrow + c * 64 + 32 * h);
                a.h[1] = *(const v8h*)(qrow + c * 64 + 32 * h + 16);
                acc = __builtin_amdgcn_wmma_f32_16x16x32_f16(false, a.v, false, kb[c],
                                                             (short)0, acc, false, false);
            }
#pragma unroll
            for (int vv = 0; vv < 8; ++vv) {
                int q = qbase + vv + 8 * h;                    // C-layout: M = v + 8h
                float s = acc[vv] * SCALE;
                float val = mcol[(size_t)q * S_] ? 1e-9f : s;  // quirk: masked -> 1e-9
                colsum += __expf(val);
            }
        }
        __syncthreads();  // LDS reads done before buffer reuse
    }
    colsum += __shfl_xor(colsum, 16, 32);
    if (h == 0) zinv[(size_t)b * S_ + kbase + ln] = 1.0f / colsum;
}

// ---------------- Pass 2: O = (exp(masked_score) * Zinv[k]) @ V ----------------
__global__ void __launch_bounds__(256)
attn_main_kernel(const _Float16* __restrict__ qh, const _Float16* __restrict__ kh,
                 const _Float16* __restrict__ vt, const uint8_t* __restrict__ mask,
                 const float* __restrict__ zinv, float* __restrict__ out) {
    // Double-buffered TDM destination for K tiles (all 8 waves share it).
    __shared__ __align__(16) unsigned char ldsK[2][BUF_BYTES];
    // Per-wave 16(q) x 32(k) f16 transpose buffer; 40-half rows keep 16B align.
    __shared__ __align__(16) _Float16 elds[8][16][40];

    const int wave = threadIdx.x >> 5;
    const int lane = threadIdx.x & 31;
    const int ln = lane & 15;
    const int h  = lane >> 4;
    const int b  = blockIdx.y;
    const int qbase = (blockIdx.x * 8 + wave) * 16;

    // Preload A fragments of Q (16 rows x 128 features, 4 chunks of K=32)
    const _Float16* qrow = qh + ((size_t)b * S_ + qbase + ln) * D_;
    HFrag qa[4];
#pragma unroll
    for (int c = 0; c < 4; ++c) {
        qa[c].h[0] = *(const v8h*)(qrow + c * 32 + 8 * h);
        qa[c].h[1] = *(const v8h*)(qrow + c * 32 + 16 + 8 * h);
    }

    v8f o[8] = {};  // 16(q) x 128(d) f32 accumulators

    if (wave == 0)  // DMA of the first K tile
        tdm_load_tile(kh + (size_t)b * S_ * D_, ldsK[0]);

    for (int kbi = 0; kbi < S_ / 32; ++kbi) {
        const int k0 = kbi * 32;
        const unsigned char* kbuf = ldsK[kbi & 1];

        __builtin_amdgcn_s_wait_tensorcnt(0);  // wave0: current tile landed
        __syncthreads();                       // publish LDS tile to all waves

        if (wave == 0 && kbi + 1 < S_ / 32)    // prefetch next K tile
            tdm_load_tile(kh + ((size_t)b * S_ + (size_t)(kbi + 1) * 32) * D_,
                          ldsK[(kbi + 1) & 1]);

        // ---- score tiles for 32 keys (two 16-key subtiles) -> exp*Zinv -> LDS
#pragma unroll
        for (int st = 0; st < 2; ++st) {
            const int kbase = k0 + st * 16;
            const unsigned char* krow = kbuf + (size_t)(st * 16 + ln) * ROW_STRIDE;
            v8f sacc = {};
#pragma unroll
            for (int c = 0; c < 4; ++c) {
                HFrag kf;  // B frag: two 16B pieces of the staged K row
                kf.h[0] = *(const v8h*)(krow + c * 64 + 16 * h);
                kf.h[1] = *(const v8h*)(krow + c * 64 + 16 * h + 32);
                sacc = __builtin_amdgcn_wmma_f32_16x16x32_f16(false, qa[c].v, false, kf.v,
                                                              (short)0, sacc, false, false);
            }
            const float zl = zinv[(size_t)b * S_ + kbase + ln];  // per-lane (N = ln)
#pragma unroll
            for (int vv = 0; vv < 8; ++vv) {
                int q = qbase + vv + 8 * h;
                float s = sacc[vv] * SCALE;
                float val = mask[((size_t)b * S_ + q) * S_ + kbase + ln] ? 1e-9f : s;
                elds[wave][vv + 8 * h][st * 16 + ln] = (_Float16)(__expf(val) * zl);
            }
        }
        __syncthreads();  // C-layout stores -> A-layout reads

        HFrag ea;  // E' in A-fragment layout (16 x 32, K=32)
        ea.h[0] = *(const v8h*)&elds[wave][ln][16 * h];
        ea.h[1] = *(const v8h*)&elds[wave][ln][16 * h + 8];

        // ---- O += E' @ V  (B frags: contiguous 32B rows of Vt[b][d][k], L1-shared)
#pragma unroll
        for (int dt = 0; dt < 8; ++dt) {
            v16h vb = *(const v16h*)(vt + ((size_t)b * D_ + dt * 16 + ln) * S_ + k0 + 16 * h);
            o[dt] = __builtin_amdgcn_wmma_f32_16x16x32_f16(false, ea.v, false, vb,
                                                           (short)0, o[dt], false, false);
        }
        __syncthreads();  // reads of ldsK/elds done before next overwrite
    }

    // ---- epilogue: C-layout -> global f32 output (half-wave coalesced rows)
#pragma unroll
    for (int dt = 0; dt < 8; ++dt) {
#pragma unroll
        for (int vv = 0; vv < 8; ++vv) {
            int q = qbase + vv + 8 * h;
            out[((size_t)b * S_ + q) * D_ + dt * 16 + ln] = o[dt][vv];
        }
    }
}

extern "C" void kernel_launch(void* const* d_in, const int* in_sizes, int n_in,
                              void* d_out, int out_size, void* d_ws, size_t ws_size,
                              hipStream_t stream) {
    (void)in_sizes; (void)n_in; (void)out_size; (void)ws_size;
    const float*   q = (const float*)d_in[0];
    const float*   k = (const float*)d_in[1];
    const float*   v = (const float*)d_in[2];
    const uint8_t* m = (const uint8_t*)d_in[3];
    float* out = (float*)d_out;

    // Workspace: Qh | Kh | Vt (f16) | Zinv (f32)  => ~25.3 MB
    const size_t BSD = (size_t)B_ * S_ * D_;
    _Float16* qh = (_Float16*)d_ws;
    _Float16* kh = qh + BSD;
    _Float16* vt = kh + BSD;
    float* zinv  = (float*)(vt + BSD);

    const int threads = 256;
    cvt_qk_kernel<<<dim3((unsigned)(BSD / threads)), threads, 0, stream>>>(q, k, qh, kh);
    cvt_vt_kernel<<<dim3((unsigned)(BSD / threads)), threads, 0, stream>>>(v, vt);
    attn_stats_kernel<<<dim3(S_ / 128, B_), threads, 0, stream>>>(qh, kh, m, zinv);
    attn_main_kernel <<<dim3(S_ / 128, B_), threads, 0, stream>>>(qh, kh, vt, m, zinv, out);
}
